// SparseMoETransformer_70257075028652
// MI455X (gfx1250) — matrix operand order repaired and checked by reading, not compiled, not run on hardware
//
#include <hip/hip_runtime.h>
#include <hip/hip_bf16.h>

// ---------------------------------------------------------------------------
// Model constants (from the reference)
// ---------------------------------------------------------------------------
#define BV   4
#define SV   512
#define DV   768
#define LV   2
#define HV   12
#define EV   8
#define HSV  64
#define FFV  3072
#define VV   32000
#define NTOK (BV * SV)          // 2048 tokens

typedef __attribute__((ext_vector_type(16))) _Float16 v16h;
typedef __attribute__((ext_vector_type(8)))  _Float16 v8h;
typedef __attribute__((ext_vector_type(4)))  _Float16 v4h;
typedef __attribute__((ext_vector_type(8)))  float    v8f;
typedef __attribute__((ext_vector_type(4)))  float    v4f;

#define GF_RELU   1
#define GF_CAUSAL 2
#define GF_ACC    4

#define LSTRIDE 72   // LDS row stride in halves (144B, 16B aligned, bank-safe)

// ---------------------------------------------------------------------------
// Generic batched-strided WMMA GEMM:  C = epilogue(alpha * A*B)
//   block = 256 threads = 8 waves; tile 128(M) x 64(N); K-step 64,
//   double-buffered LDS; wave tile 32x32 = 2x2 v_wmma_f32_16x16x32_f16,
//   8 WMMAs per K-step per wave.
// ---------------------------------------------------------------------------
__global__ __launch_bounds__(256)
void gemm_wmma_f16(const float* __restrict__ A, int lda, long sAo, long sAi,
                   const float* __restrict__ B, int ldb, long sBo, long sBi,
                   int transB,
                   float* __restrict__ C, int ldc, long sCo, long sCi,
                   const float* __restrict__ bias, long sBiasI,
                   const float* __restrict__ rowscale, int rsStride,
                   int M, int N, int Kd, int zInner,
                   float alpha, int flags)
{
    __shared__ alignas(16) _Float16 lA[2][128 * LSTRIDE];  // 2 x 18 KB
    __shared__ alignas(16) _Float16 lB[2][64 * LSTRIDE];   // 2 x  9 KB

    const int z  = blockIdx.z;
    const int zo = z / zInner;
    const int zi = z % zInner;
    A += zo * sAo + zi * sAi;
    B += zo * sBo + zi * sBi;
    C += zo * sCo + zi * sCi;
    if (bias) bias += zi * sBiasI;

    const int rowBase = blockIdx.y * 128;
    const int colBase = blockIdx.x * 64;

    const int tid   = threadIdx.x;
    const int wave  = tid >> 5;
    const int lane  = tid & 31;
    const int wr    = wave >> 1;        // 0..3 : 32-row group
    const int wc    = wave & 1;         // 0..1 : 32-col group
    const int lm    = lane & 15;
    const int khalf = lane >> 4;        // 0/1

    // staging coordinates: each 64-half row is 16 x v4f; 256 threads
    const int sr  = tid >> 4;           // 0..15
    const int sc4 = tid & 15;           // 0..15

    const float* aP = A + (long)(rowBase + sr) * lda + sc4 * 4;
    const float* bPt = B + (long)(colBase + sr) * ldb + sc4 * 4;  // transB
    const float* bPn = B + (long)sr * ldb + colBase + sc4 * 4;    // !transB

    const int nK = Kd >> 6;             // number of 64-wide K steps

    v8f acc[2][2] = {};
    v4f ra[8], rb[4];

    // ---- fully unrolled global loads for K-step kt (into registers) ----
    auto loadTile = [&](int kt) {
        const float* ap = aP + (long)kt * 64;
#pragma unroll
        for (int i = 0; i < 8; ++i)
            ra[i] = *(const v4f*)(ap + (long)i * 16 * lda);
        if (transB) {
            const float* bp = bPt + (long)kt * 64;
#pragma unroll
            for (int i = 0; i < 4; ++i)
                rb[i] = *(const v4f*)(bp + (long)i * 16 * ldb);
        } else {
            const float* bp = bPn + (long)kt * 64 * ldb;
#pragma unroll
            for (int i = 0; i < 4; ++i)
                rb[i] = *(const v4f*)(bp + (long)i * 16 * ldb);
        }
    };

    // ---- convert + store staged registers into LDS buffer `buf` ----
    auto storeTile = [&](int buf) {
        _Float16* la = &lA[buf][0] + sr * LSTRIDE + sc4 * 4;
#pragma unroll
        for (int i = 0; i < 8; ++i)
            *(v4h*)(la + i * 16 * LSTRIDE) = __builtin_convertvector(ra[i], v4h);
        if (transB) {
            _Float16* lb = &lB[buf][0] + sr * LSTRIDE + sc4 * 4;
#pragma unroll
            for (int i = 0; i < 4; ++i)
                *(v4h*)(lb + i * 16 * LSTRIDE) = __builtin_convertvector(rb[i], v4h);
        } else {
            // rb[i] = B[k = sr + 16*i][n = colBase + 4*sc4 + u]
            _Float16* lb = &lB[buf][0];
#pragma unroll
            for (int i = 0; i < 4; ++i) {
                int k = sr + i * 16;
#pragma unroll
                for (int u = 0; u < 4; ++u)
                    lb[(sc4 * 4 + u) * LSTRIDE + k] = (_Float16)rb[i][u];
            }
        }
    };

    // ---- 8 WMMAs on LDS buffer `buf` (two k-subs of 32) ----
    auto compute = [&](int buf) {
        const _Float16* la = &lA[buf][0];
        const _Float16* lb = &lB[buf][0];
#pragma unroll
        for (int ks = 0; ks < 64; ks += 32) {
            v16h af[2], bf[2];
#pragma unroll
            for (int i = 0; i < 2; ++i) {
                const _Float16* pa = la + (wr * 32 + i * 16 + lm) * LSTRIDE + ks;
                v8h lo = *(const v8h*)(pa + khalf * 8);
                v8h hi = *(const v8h*)(pa + 16 + khalf * 8);
                af[i] = __builtin_shufflevector(lo, hi, 0,1,2,3,4,5,6,7,
                                                        8,9,10,11,12,13,14,15);
            }
#pragma unroll
            for (int j = 0; j < 2; ++j) {
                const _Float16* pb = lb + (wc * 32 + j * 16 + lm) * LSTRIDE
                                        + ks;
                v8h lo = *(const v8h*)(pb + khalf * 8);
                v8h hi = *(const v8h*)(pb + 16 + khalf * 8);
                bf[j] = __builtin_shufflevector(lo, hi, 0,1,2,3,4,5,6,7,
                                                        8,9,10,11,12,13,14,15);
            }
#pragma unroll
            for (int i = 0; i < 2; ++i)
#pragma unroll
                for (int j = 0; j < 2; ++j)
                    acc[i][j] = __builtin_amdgcn_wmma_f32_16x16x32_f16(
                        false, af[i], false, bf[j], (short)0, acc[i][j],
                        false, false);
        }
    };

    // ---- software pipeline: prefetch next tile during WMMA ----
    loadTile(0);
    storeTile(0);
    __syncthreads();
    for (int kt = 0; kt < nK; ++kt) {
        const int cur = kt & 1;
        if (kt + 1 < nK) loadTile(kt + 1);   // global loads in flight ...
        compute(cur);                        // ... across the WMMA section
        if (kt + 1 < nK) storeTile(cur ^ 1);
        __syncthreads();
    }

    // ---- epilogue: alpha, bias, relu, causal mask, row scale, accumulate
#pragma unroll
    for (int i = 0; i < 2; ++i) {
#pragma unroll
        for (int j = 0; j < 2; ++j) {
            int nCol = colBase + wc * 32 + j * 16 + lm;
#pragma unroll
            for (int r = 0; r < 8; ++r) {
                int row = rowBase + wr * 32 + i * 16 + khalf * 8 + r;
                float v = acc[i][j][r] * alpha;
                if (bias) v += bias[nCol];
                if (flags & GF_RELU) v = v > 0.f ? v : 0.f;
                if ((flags & GF_CAUSAL) && nCol > row) v = -__builtin_inff();
                if (rowscale) v *= rowscale[(long)row * rsStride];
                float* cp = C + (long)row * ldc + nCol;
                if (flags & GF_ACC) v += *cp;
                *cp = v;
            }
        }
    }
}

// ---------------------------------------------------------------------------
// Embedding: x = tok_emb[tokens] + pos_emb
// ---------------------------------------------------------------------------
__global__ __launch_bounds__(256)
void embed_k(const int* __restrict__ tokens, const float* __restrict__ tokEmb,
             const float* __restrict__ posEmb, float* __restrict__ x)
{
    long i = (long)blockIdx.x * 256 + threadIdx.x;   // over NTOK*DV
    if (i >= (long)NTOK * DV) return;
    int d  = (int)(i % DV);
    int ts = (int)(i / DV);
    int s  = ts % SV;
    int tok = tokens[ts];
    x[i] = tokEmb[(long)tok * DV + d] + posEmb[(long)s * DV + d];
}

// ---------------------------------------------------------------------------
// Row LayerNorm (one 256-thread block per row)
// ---------------------------------------------------------------------------
__global__ __launch_bounds__(256)
void layernorm_k(const float* __restrict__ x, const float* __restrict__ g,
                 const float* __restrict__ b, float* __restrict__ out, int Dd)
{
    __shared__ float red[256];
    const int tid = threadIdx.x;
    const float* xr = x + (long)blockIdx.x * Dd;
    float* orow = out + (long)blockIdx.x * Dd;

    float s = 0.f;
    for (int i = tid; i < Dd; i += 256) s += xr[i];
    red[tid] = s; __syncthreads();
    for (int off = 128; off > 0; off >>= 1) {
        if (tid < off) red[tid] += red[tid + off];
        __syncthreads();
    }
    float mu = red[0] / (float)Dd;
    __syncthreads();

    float v = 0.f;
    for (int i = tid; i < Dd; i += 256) { float d = xr[i] - mu; v += d * d; }
    red[tid] = v; __syncthreads();
    for (int off = 128; off > 0; off >>= 1) {
        if (tid < off) red[tid] += red[tid + off];
        __syncthreads();
    }
    float rstd = rsqrtf(red[0] / (float)Dd + 1e-5f);

    for (int i = tid; i < Dd; i += 256)
        orow[i] = (xr[i] - mu) * rstd * g[i] + b[i];
}

// ---------------------------------------------------------------------------
// Row softmax in place (handles -inf masked entries)
// ---------------------------------------------------------------------------
__global__ __launch_bounds__(256)
void softmax_k(float* __restrict__ s, int N)
{
    __shared__ float red[256];
    const int tid = threadIdx.x;
    float* row = s + (long)blockIdx.x * N;

    float m = -__builtin_inff();
    for (int i = tid; i < N; i += 256) m = fmaxf(m, row[i]);
    red[tid] = m; __syncthreads();
    for (int off = 128; off > 0; off >>= 1) {
        if (tid < off) red[tid] = fmaxf(red[tid], red[tid + off]);
        __syncthreads();
    }
    float mm = red[0];
    __syncthreads();

    float sum = 0.f;
    for (int i = tid; i < N; i += 256) {
        float e = __expf(row[i] - mm);
        row[i] = e;
        sum += e;
    }
    red[tid] = sum; __syncthreads();
    for (int off = 128; off > 0; off >>= 1) {
        if (tid < off) red[tid] += red[tid + off];
        __syncthreads();
    }
    float inv = 1.f / red[0];
    for (int i = tid; i < N; i += 256) row[i] *= inv;
}

// ---------------------------------------------------------------------------
// Router scores: sc[t,e] = y[t,:] . Wr[:,e] + br[e]   (tiny: plain VALU)
// ---------------------------------------------------------------------------
__global__ __launch_bounds__(256)
void router_k(const float* __restrict__ y, const float* __restrict__ Wr,
              const float* __restrict__ br, float* __restrict__ sc)
{
    int gid = blockIdx.x * 256 + threadIdx.x;   // NTOK*EV
    if (gid >= NTOK * EV) return;
    int t = gid >> 3, e = gid & 7;
    const float* yr = y + (long)t * DV;
    float s = br[e];
    for (int d = 0; d < DV; ++d) s += yr[d] * Wr[d * EV + e];
    sc[gid] = s;
}

// ---------------------------------------------------------------------------
// Top-2 gating: probs[t,:] = scatter(softmax(top2(sc[t,:])))
// ---------------------------------------------------------------------------
__global__ __launch_bounds__(256)
void top2_k(const float* __restrict__ sc, float* __restrict__ probs)
{
    int t = blockIdx.x * 256 + threadIdx.x;
    if (t >= NTOK) return;
    const float* s = sc + (long)t * EV;
    int i1 = 0; float v1 = s[0];
    for (int e = 1; e < EV; ++e) if (s[e] > v1) { v1 = s[e]; i1 = e; }
    int i2 = -1; float v2 = -3.4e38f;
    for (int e = 0; e < EV; ++e)
        if (e != i1 && s[e] > v2) { v2 = s[e]; i2 = e; }
    float e2 = __expf(v2 - v1);           // v1 is the max
    float inv = 1.f / (1.f + e2);
    float* p = probs + (long)t * EV;
    for (int e = 0; e < EV; ++e) p[e] = 0.f;
    p[i1] = inv;
    p[i2] = e2 * inv;
}

// ---------------------------------------------------------------------------
// Host-side orchestration
// ---------------------------------------------------------------------------
extern "C" void kernel_launch(void* const* d_in, const int* in_sizes, int n_in,
                              void* d_out, int out_size, void* d_ws, size_t ws_size,
                              hipStream_t stream)
{
    (void)in_sizes; (void)n_in; (void)out_size; (void)ws_size;

    const int*   tokens  = (const int*)  d_in[0];
    const float* tok_emb = (const float*)d_in[1];
    const float* pos_emb = (const float*)d_in[2];
    const float* ln1_g   = (const float*)d_in[3];
    const float* ln1_b   = (const float*)d_in[4];
    const float* Wq      = (const float*)d_in[5];
    const float* bq      = (const float*)d_in[6];
    const float* Wk      = (const float*)d_in[7];
    const float* bk      = (const float*)d_in[8];
    const float* Wv      = (const float*)d_in[9];
    const float* bv      = (const float*)d_in[10];
    const float* Wo      = (const float*)d_in[11];
    const float* bo      = (const float*)d_in[12];
    const float* ln2_g   = (const float*)d_in[13];
    const float* ln2_b   = (const float*)d_in[14];
    const float* Wr      = (const float*)d_in[15];
    const float* br      = (const float*)d_in[16];
    const float* W1      = (const float*)d_in[17];
    const float* b1      = (const float*)d_in[18];
    const float* W2      = (const float*)d_in[19];
    const float* b2      = (const float*)d_in[20];
    const float* lnf_g   = (const float*)d_in[21];
    const float* lnf_b   = (const float*)d_in[22];
    const float* Wout    = (const float*)d_in[23];
    const float* bout    = (const float*)d_in[24];

    // ---- workspace layout (bytes) ----
    const long SZ_TD = (long)NTOK * DV * 4;            // 6,291,456
    char* ws = (char*)d_ws;
    float* X  = (float*)(ws + 0 * SZ_TD);              // residual stream
    float* LN = (float*)(ws + 1 * SZ_TD);              // layernorm output
    float* Qb = (float*)(ws + 2 * SZ_TD);
    float* Kb = (float*)(ws + 3 * SZ_TD);
    float* Vb = (float*)(ws + 4 * SZ_TD);
    float* AO = (float*)(ws + 5 * SZ_TD);              // attention output
    float* Sc = (float*)(ws + 6 * SZ_TD);              // scores 48*512*512 f32
    float* HB = Sc;                                    // MoE hidden aliases scores
    long offRS = 6 * SZ_TD + (long)BV * HV * SV * SV * 4;
    float* RS = (float*)(ws + offRS);                  // router scores 2048x8
    float* Pb = (float*)(ws + offRS + (long)NTOK * EV * 4);  // gate probs 2048x8

    const dim3 blk(256);
    const float inv_sqrt_d = 0.03608439182435161f;     // 1/sqrt(768)

    // ---- embedding ----
    embed_k<<<dim3((NTOK * DV) / 256), blk, 0, stream>>>(tokens, tok_emb, pos_emb, X);

    for (int l = 0; l < LV; ++l) {
        // ---- LN1 ----
        layernorm_k<<<dim3(NTOK), blk, 0, stream>>>(X, ln1_g + l * DV, ln1_b + l * DV, LN, DV);

        // ---- Q,K,V projections: per-head GEMM (z over heads) ----
        const float* Wqkv[3] = { Wq + (long)l * HV * DV * HSV,
                                 Wk + (long)l * HV * DV * HSV,
                                 Wv + (long)l * HV * DV * HSV };
        const float* bqkv[3] = { bq + (long)l * HV * HSV,
                                 bk + (long)l * HV * HSV,
                                 bv + (long)l * HV * HSV };
        float* outqkv[3] = { Qb, Kb, Vb };
        for (int m = 0; m < 3; ++m) {
            gemm_wmma_f16<<<dim3(1, NTOK / 128, HV), blk, 0, stream>>>(
                LN, DV, 0, 0,
                Wqkv[m], HSV, 0, (long)DV * HSV, 0,
                outqkv[m], DV, 0, HSV,
                bqkv[m], HSV,
                nullptr, 0,
                NTOK, HSV, DV, HV, 1.0f, 0);
        }

        // ---- scores = Q K^T / sqrt(D), causal mask (z over b*H+h) ----
        gemm_wmma_f16<<<dim3(SV / 64, SV / 128, BV * HV), blk, 0, stream>>>(
            Qb, DV, (long)SV * DV, HSV,
            Kb, DV, (long)SV * DV, HSV, 1,
            Sc, SV, (long)HV * SV * SV, (long)SV * SV,
            nullptr, 0,
            nullptr, 0,
            SV, SV, HSV, HV, inv_sqrt_d, GF_CAUSAL);

        // ---- softmax over rows ----
        softmax_k<<<dim3(BV * HV * SV), blk, 0, stream>>>(Sc, SV);

        // ---- out = P V  ->  AO laid out (b,s,h*64+e) ----
        gemm_wmma_f16<<<dim3(1, SV / 128, BV * HV), blk, 0, stream>>>(
            Sc, SV, (long)HV * SV * SV, (long)SV * SV,
            Vb, DV, (long)SV * DV, HSV, 0,
            AO, DV, (long)SV * DV, HSV,
            nullptr, 0,
            nullptr, 0,
            SV, HSV, SV, HV, 1.0f, 0);

        // ---- x += AO @ Wo + bo ----
        gemm_wmma_f16<<<dim3(DV / 64, NTOK / 128, 1), blk, 0, stream>>>(
            AO, DV, 0, 0,
            Wo + (long)l * DV * DV, DV, 0, 0, 0,
            X, DV, 0, 0,
            bo + (long)l * DV, 0,
            nullptr, 0,
            NTOK, DV, DV, 1, 1.0f, GF_ACC);

        // ---- LN2 -> y ----
        layernorm_k<<<dim3(NTOK), blk, 0, stream>>>(X, ln2_g + l * DV, ln2_b + l * DV, LN, DV);

        // ---- router + top-2 gating ----
        router_k<<<dim3((NTOK * EV) / 256), blk, 0, stream>>>(
            LN, Wr + (long)l * DV * EV, br + (long)l * EV, RS);
        top2_k<<<dim3(NTOK / 256), blk, 0, stream>>>(RS, Pb);

        // ---- dense experts: x += prob_e * (relu(y W1 + b1) W2 + b2) ----
        for (int e = 0; e < EV; ++e) {
            const long we = (long)l * EV + e;
            gemm_wmma_f16<<<dim3(FFV / 64, NTOK / 128, 1), blk, 0, stream>>>(
                LN, DV, 0, 0,
                W1 + we * DV * FFV, FFV, 0, 0, 0,
                HB, FFV, 0, 0,
                b1 + we * FFV, 0,
                nullptr, 0,
                NTOK, FFV, DV, 1, 1.0f, GF_RELU);

            gemm_wmma_f16<<<dim3(DV / 64, NTOK / 128, 1), blk, 0, stream>>>(
                HB, FFV, 0, 0,
                W2 + we * FFV * DV, DV, 0, 0, 0,
                X, DV, 0, 0,
                b2 + we * DV, 0,
                Pb + e, EV,
                NTOK, DV, FFV, 1, 1.0f, GF_ACC);
        }
    }

    // ---- final LN + vocab projection ----
    layernorm_k<<<dim3(NTOK), blk, 0, stream>>>(X, lnf_g, lnf_b, LN, DV);
    gemm_wmma_f16<<<dim3(VV / 64, NTOK / 128, 1), blk, 0, stream>>>(
        LN, DV, 0, 0,
        Wout, VV, 0, 0, 0,
        (float*)d_out, VV, 0, 0,
        bout, 0,
        nullptr, 0,
        NTOK, VV, DV, 1, 1.0f, 0);
}
